// NADE_16209206575807
// MI455X (gfx1250) — matrix operand order repaired and checked by reading, not compiled, not run on hardware
//
#include <hip/hip_runtime.h>

// Problem sizes (fixed by reference)
#define NN 4096   // visible units
#define HH 2048   // hidden units
#define BB 64     // batch
#define NCHUNK 64 // chunks
#define CS 64     // steps per chunk (NCHUNK*CS == NN)

typedef __attribute__((ext_vector_type(16))) _Float16 v16h;
typedef __attribute__((ext_vector_type(2)))  _Float16 h2;
typedef __attribute__((ext_vector_type(8)))  float    v8f;

// Fast sigmoid: prefer the gfx1250 v_tanh_f32 transcendental (1 trans op)
// over exp+rcp (2 trans ops).  sigma(z) = 0.5*tanh(z/2) + 0.5.
__device__ __forceinline__ float fast_sigmoid(float z) {
#if __has_builtin(__builtin_amdgcn_tanhf)
  return fmaf(0.5f, __builtin_amdgcn_tanhf(0.5f * z), 0.5f);
#else
  return __builtin_amdgcn_rcpf(1.f + __expf(-z));
#endif
}

// ---------------------------------------------------------------------------
// Kernel P: transpose W (HH x NN, f32) -> Wt (NN x HH, f16) for coalesced
// per-step column access in the scan kernel.
// ---------------------------------------------------------------------------
__global__ __launch_bounds__(256) void nade_transpose(
    const float* __restrict__ W, _Float16* __restrict__ Wt) {
  __shared__ float tile[32][33];
  const int i0 = blockIdx.x * 32;  // column base in W (NN dim)
  const int k0 = blockIdx.y * 32;  // row base in W (HH dim)
  const int tx = threadIdx.x;      // 0..31
  const int ty = threadIdx.y;      // 0..7
#pragma unroll
  for (int s = 0; s < 4; ++s) {
    const int k = k0 + ty + 8 * s;
    tile[ty + 8 * s][tx] = W[(size_t)k * NN + i0 + tx];
  }
  __syncthreads();
#pragma unroll
  for (int s = 0; s < 4; ++s) {
    const int i = i0 + ty + 8 * s;
    Wt[(size_t)i * HH + k0 + tx] = (_Float16)tile[tx][ty + 8 * s];
  }
}

// ---------------------------------------------------------------------------
// Kernel A: per-chunk rank-CS sums via WMMA.
//   S[c][k][b] = sum_{j<CS} W[k, c*CS+j] * x[b, c*CS+j]
// One wave per (chunk, hidden-group of 16 tiles). B operand (x chunk, 4 batch
// tiles) is held in registers and reused across 16 hidden tiles; 8 WMMAs/tile.
// ---------------------------------------------------------------------------
__global__ __launch_bounds__(32) void nade_chunk_gemm(
    const float* __restrict__ W, const float* __restrict__ x,
    float* __restrict__ S) {
  const int c    = blockIdx.x;       // chunk
  const int g    = blockIdx.y;       // hidden group: 16 tiles of 16 rows
  const int lane = threadIdx.x;      // 0..31
  const int half = lane >> 4;        // 0/1
  const int lid  = lane & 15;
  const int i0   = c * CS;

  // B operand: B[K=j, n=b] = x[b, i0+j].  Lane = n, elements 0-7 <->
  // K = 8*half + 0..7, elements 8-15 <-> K += 16 (matches A).
  v16h Bmat[4][2];
#pragma unroll
  for (int bt = 0; bt < 4; ++bt) {
    const int bb = bt * 16 + lid;
#pragma unroll
    for (int kb = 0; kb < 2; ++kb) {
      const float* xp = x + (size_t)bb * NN + i0 + kb * 32;
      v16h t;
#pragma unroll
      for (int e = 0; e < 8; ++e) t[e] = (_Float16)xp[half * 8 + e];
#pragma unroll
      for (int e = 0; e < 8; ++e) t[8 + e] = (_Float16)xp[16 + half * 8 + e];
      Bmat[bt][kb] = t;
    }
  }

  for (int tIdx = 0; tIdx < 16; ++tIdx) {
    const int m = (g * 16 + tIdx) * 16 + lid;  // hidden row (lane = M)
    // A 16x32 f16: VGPRs 0-3 <-> K = 8*half + 0..7, VGPRs 4-7 <-> K += 16.
    v16h Amat[2];
#pragma unroll
    for (int kb = 0; kb < 2; ++kb) {
      const float* wp = W + (size_t)m * NN + i0 + kb * 32;
      v16h t;
#pragma unroll
      for (int e = 0; e < 8; ++e) t[e] = (_Float16)wp[half * 8 + e];
#pragma unroll
      for (int e = 0; e < 8; ++e) t[8 + e] = (_Float16)wp[16 + half * 8 + e];
      Amat[kb] = t;
    }
#pragma unroll
    for (int bt = 0; bt < 4; ++bt) {
      v8f d = {};
      d = __builtin_amdgcn_wmma_f32_16x16x32_f16(
              false, Amat[0], false, Bmat[bt][0], (short)0, d, false, false);
      d = __builtin_amdgcn_wmma_f32_16x16x32_f16(
              false, Amat[1], false, Bmat[bt][1], (short)0, d, false, false);
      // D: lane = n, VGPR r -> row m0 + r + 8*half
      float* sp = S + ((size_t)c * HH + (g * 16 + tIdx) * 16 + 8 * half) * BB
                    + bt * 16 + lid;
#pragma unroll
      for (int r = 0; r < 8; ++r) sp[(size_t)r * BB] = d[r];
    }
  }
}

// ---------------------------------------------------------------------------
// Kernel B: in-place exclusive prefix over chunks, folding in hidden bias c:
//   A0[c][k][b] = cbias[k] + sum_{c'<c} S[c'][k][b]
// Each thread owns one (k,b) column -> deterministic, no races.
// ---------------------------------------------------------------------------
__global__ __launch_bounds__(256) void nade_prefix(
    const float* __restrict__ cbias, float* __restrict__ S) {
  const int t = blockIdx.x * blockDim.x + threadIdx.x;  // 0 .. HH*BB-1
  const int k = t >> 6;
  float run = cbias[k];
  float* p = S + t;
  for (int c = 0; c < NCHUNK; ++c) {
    const float s = p[(size_t)c * HH * BB];
    p[(size_t)c * HH * BB] = run;
    run += s;
  }
}

// ---------------------------------------------------------------------------
// Kernel C: intra-chunk sequential scan. One wave per (chunk, batch row).
// Accumulator (c + a) for the full hidden dim lives in registers:
// lane owns k = 64*j + 2*lane + {0,1}, j = 0..31  (coalesced V/Wt streaming).
// x is binary and wave-uniform: when x==0 the accumulator update (and the
// Wt load feeding it) is skipped entirely via a wave-uniform branch.
// ---------------------------------------------------------------------------
__global__ __launch_bounds__(256) void nade_scan(
    const float* __restrict__ x, const float* __restrict__ V,
    const float* __restrict__ bbias, const _Float16* __restrict__ Wt,
    const float* __restrict__ A0, float* __restrict__ partials) {
  const int c    = blockIdx.x;
  const int wave = threadIdx.x >> 5;
  const int lane = threadIdx.x & 31;
  const int b    = blockIdx.y * 8 + wave;

  // Load chunk-start accumulator (includes hidden bias).
  float2 acc[32];
  const float* a0p = A0 + (size_t)c * HH * BB + b;
#pragma unroll
  for (int j = 0; j < 32; ++j) {
    const int k = 64 * j + 2 * lane;
    acc[j].x = a0p[(size_t)k * BB];
    acc[j].y = a0p[(size_t)(k + 1) * BB];
  }

  float logp = 0.f;
  const float* xrow = x + (size_t)b * NN + c * CS;

#pragma unroll 1
  for (int js = 0; js < CS; ++js) {
    const int i = c * CS + js;
    const float xv = xrow[js];
    const bool xset = (xv > 0.5f);  // wave-uniform
    const float2* v2 = (const float2*)(V + (size_t)i * HH);
    const h2*     w2 = (const h2*)(Wt + (size_t)i * HH);
    // Prefetch next step's rows (global_prefetch_b8); streams from L2.
    const int inext = (js < CS - 1) ? i + 1 : i;
    __builtin_prefetch((const void*)(V + (size_t)inext * HH + 2 * lane), 0, 1);
    __builtin_prefetch((const void*)(Wt + (size_t)inext * HH + 2 * lane), 0, 1);

    float dot = 0.f;
    if (xset) {
      // x==1: sigmoid + dot + accumulator update (Wt streamed).
#pragma unroll
      for (int j = 0; j < 32; ++j) {
        const int kp = 32 * j + lane;  // pair index
        const float2 vv = v2[kp];
        const h2     wv = w2[kp];
        const float z0 = acc[j].x, z1 = acc[j].y;
        const float h0 = fast_sigmoid(z0);
        const float h1 = fast_sigmoid(z1);
        dot = fmaf(h0, vv.x, dot);
        dot = fmaf(h1, vv.y, dot);
        acc[j].x = z0 + (float)wv[0];
        acc[j].y = z1 + (float)wv[1];
      }
    } else {
      // x==0: accumulator unchanged; skip Wt entirely.
#pragma unroll
      for (int j = 0; j < 32; ++j) {
        const int kp = 32 * j + lane;
        const float2 vv = v2[kp];
        const float h0 = fast_sigmoid(acc[j].x);
        const float h1 = fast_sigmoid(acc[j].y);
        dot = fmaf(h0, vv.x, dot);
        dot = fmaf(h1, vv.y, dot);
      }
    }
    // Wave32 butterfly reduction of the dot product.
#pragma unroll
    for (int off = 16; off > 0; off >>= 1) dot += __shfl_xor(dot, off, 32);

    const float logit = dot + bbias[i];
    const float u = xset ? logit : -logit;  // log sigmoid(u), stable form
    logp += fminf(u, 0.f) - __logf(1.f + __expf(-fabsf(u)));
  }
  if (lane == 0) partials[(size_t)b * NCHUNK + c] = logp;
}

// ---------------------------------------------------------------------------
// Kernel D: deterministic final reduction (fixed order, no atomics).
// ---------------------------------------------------------------------------
__global__ __launch_bounds__(64) void nade_finish(
    const float* __restrict__ partials, float* __restrict__ out) {
  const int b = threadIdx.x;
  if (b < BB) {
    float s = 0.f;
    for (int c = 0; c < NCHUNK; ++c) s += partials[(size_t)b * NCHUNK + c];
    out[b] = s;
  }
}

// ---------------------------------------------------------------------------
extern "C" void kernel_launch(void* const* d_in, const int* in_sizes, int n_in,
                              void* d_out, int out_size, void* d_ws,
                              size_t ws_size, hipStream_t stream) {
  (void)in_sizes; (void)n_in; (void)out_size; (void)ws_size;
  const float* x  = (const float*)d_in[0];  // (64, 4096)
  const float* W  = (const float*)d_in[1];  // (2048, 4096)
  const float* cb = (const float*)d_in[2];  // (2048,)
  const float* V  = (const float*)d_in[3];  // (4096, 2048)
  const float* bb = (const float*)d_in[4];  // (4096,)
  float* out = (float*)d_out;               // (64,)

  char* ws = (char*)d_ws;
  constexpr size_t WT_BYTES = (size_t)NN * HH * 2;            // 16 MiB
  constexpr size_t S_BYTES  = (size_t)NCHUNK * HH * BB * 4;   // 32 MiB
  _Float16* Wt      = (_Float16*)ws;
  float*    S       = (float*)(ws + WT_BYTES);
  float*    partial = (float*)(ws + WT_BYTES + S_BYTES);

  nade_transpose<<<dim3(NN / 32, HH / 32), dim3(32, 8), 0, stream>>>(W, Wt);
  nade_chunk_gemm<<<dim3(NCHUNK, 8), 32, 0, stream>>>(W, x, S);
  nade_prefix<<<(HH * BB) / 256, 256, 0, stream>>>(cb, S);
  nade_scan<<<dim3(NCHUNK, BB / 8), 256, 0, stream>>>(x, V, bb, Wt, S, partial);
  nade_finish<<<1, 64, 0, stream>>>(partial, out);
}
